// RoiMultiHead_56538949484855
// MI455X (gfx1250) — matrix operand order repaired and checked by reading, not compiled, not run on hardware
//
#include <hip/hip_runtime.h>

// ---------------------------------------------------------------------------
// RoiMultiHead on MI455X (gfx1250, wave32, WMMA)
// reg = feats @ W_reg[roi]^T + b_reg[roi]   -> [B, 6]
// cls = feats @ W_cls[roi]^T + b_cls[roi]   -> [B, 3, 16]
// Strategy: bucket-by-roi counting sort, then grouped GEMM with
// v_wmma_f32_16x16x32_bf16 using hi/lo bf16 error-compensated split
// (Ahi*Bhi + Ahi*Blo + Alo*Bhi ~= fp32 accuracy).
// ---------------------------------------------------------------------------

#define B_ 16384
#define D_ 2048
#define G_ 8
#define R_ 6
#define KC_ 48           // K*C = 3*16
#define NPAD_ 64         // padded N: rows 0..5 reg, 16..63 cls

typedef __attribute__((ext_vector_type(16))) __bf16 v16bf;
typedef __attribute__((ext_vector_type(8)))  __bf16 v8bf;
typedef __attribute__((ext_vector_type(8)))  float  v8f;

// ---- workspace layout (bytes) ----
#define WS_COUNTS   0                         // 8 ints
#define WS_ORDER    256                       // G_*B_ ints = 512 KB
#define WS_WHI      (256 + G_*B_*4)           // G_*NPAD_*D_ bf16 = 2 MB (32B aligned)
#define WS_WLO      (WS_WHI + G_*NPAD_*D_*2)  // G_*NPAD_*D_ bf16 = 2 MB

__global__ void roi_zero_counts(int* __restrict__ counts) {
    if (threadIdx.x < G_) counts[threadIdx.x] = 0;
}

__global__ void roi_bucket(const int* __restrict__ roi_ids,
                           int* __restrict__ counts,
                           int* __restrict__ order) {
    int i = blockIdx.x * blockDim.x + threadIdx.x;
    if (i >= B_) return;
    int g = roi_ids[i];
    int pos = atomicAdd(&counts[g], 1);
    order[g * B_ + pos] = i;
}

// Pack W_reg (rows 0..5) and W_cls (rows 16..63) per group into padded
// [G_][NPAD_][D_] bf16 hi/lo matrices. Padded rows 6..15 are stored as zeros.
__global__ void roi_convert_w(const float* __restrict__ W_reg,
                              const float* __restrict__ W_cls,
                              __bf16* __restrict__ Whi,
                              __bf16* __restrict__ Wlo) {
    int idx = blockIdx.x * blockDim.x + threadIdx.x;   // over G_*NPAD_*D_
    int k   = idx % D_;
    int row = (idx / D_) % NPAD_;
    int g   = idx / (D_ * NPAD_);
    float x = 0.0f;
    if (row < R_)        x = W_reg[((size_t)g * R_ + row) * D_ + k];
    else if (row >= 16)  x = W_cls[((size_t)g * KC_ + (row - 16)) * D_ + k];
    __bf16 hi = (__bf16)x;
    __bf16 lo = (__bf16)(x - (float)hi);
    Whi[idx] = hi;
    Wlo[idx] = lo;
}

// One block = 16 gathered samples of group g x all 64 padded outputs.
// wave 0 -> reg tile (cols 0..5 valid), waves 1..3 -> cls tiles.
__global__ __launch_bounds__(128) void roi_gemm(
    const float* __restrict__ feats,
    const int*   __restrict__ counts,
    const int*   __restrict__ order,
    const __bf16* __restrict__ Whi,
    const __bf16* __restrict__ Wlo,
    const float* __restrict__ b_reg,
    const float* __restrict__ b_cls,
    float* __restrict__ out)
{
    __shared__ __bf16 ldsAhi[16 * 32];
    __shared__ __bf16 ldsAlo[16 * 32];
    __shared__ int    s_rows[16];

    const int g = blockIdx.x >> 10;            // 1024 tile slots per group
    const int t = blockIdx.x & 1023;
    const int cnt = counts[g];
    if (t * 16 >= cnt) return;                 // uniform early exit

    const int tid  = threadIdx.x;
    const int wave = tid >> 5;                 // n-tile id 0..3
    const int lane = tid & 31;
    const int n    = lane & 15;                // output column within tile
    const int h    = lane >> 4;                // lane-half

    if (tid < 16) {
        int idx = t * 16 + tid;
        s_rows[tid] = (idx < cnt) ? order[g * B_ + idx] : order[g * B_];
    }
    __syncthreads();

    // cooperative A staging: thread -> (row = tid/8, 4-float chunk = tid%8)
    const int arow = tid >> 3;
    const int achk = tid & 7;
    const float* frow = feats + (size_t)s_rows[arow] * D_ + achk * 4;
    const int    lbase = arow * 32 + achk * 4;

    // per-lane B row pointers (padded weight matrix, bf16 hi/lo)
    const size_t wrow = ((size_t)g * NPAD_ + wave * 16 + n) * D_ + 16 * h;
    const __bf16* whp = Whi + wrow;
    const __bf16* wlp = Wlo + wrow;

    // per-lane A fragment addresses in LDS (m = lane&15)
    const __bf16* pah0 = &ldsAhi[(lane & 15) * 32 + 16 * h];
    const __bf16* pah1 = &ldsAhi[(lane & 15) * 32 + 16 * h + 8];
    const __bf16* pal0 = &ldsAlo[(lane & 15) * 32 + 16 * h];
    const __bf16* pal1 = &ldsAlo[(lane & 15) * 32 + 16 * h + 8];

    v8f acc = {};
    for (int kb = 0; kb < D_; kb += 32) {
        // stage 16x32 fp32 slab -> bf16 hi/lo in LDS
        float4 x = *(const float4*)(frow + kb);
        __bf16 h0 = (__bf16)x.x, h1 = (__bf16)x.y,
               h2 = (__bf16)x.z, h3 = (__bf16)x.w;
        ldsAhi[lbase + 0] = h0; ldsAhi[lbase + 1] = h1;
        ldsAhi[lbase + 2] = h2; ldsAhi[lbase + 3] = h3;
        ldsAlo[lbase + 0] = (__bf16)(x.x - (float)h0);
        ldsAlo[lbase + 1] = (__bf16)(x.y - (float)h1);
        ldsAlo[lbase + 2] = (__bf16)(x.z - (float)h2);
        ldsAlo[lbase + 3] = (__bf16)(x.w - (float)h3);
        __syncthreads();

        v8bf a0 = *(const v8bf*)(pah0 + 0);
        v8bf a1 = *(const v8bf*)(pah1 + 0);
        v8bf l0 = *(const v8bf*)(pal0 + 0);
        v8bf l1 = *(const v8bf*)(pal1 + 0);
        v16bf ah = __builtin_shufflevector(a0, a1, 0,1,2,3,4,5,6,7,8,9,10,11,12,13,14,15);
        v16bf al = __builtin_shufflevector(l0, l1, 0,1,2,3,4,5,6,7,8,9,10,11,12,13,14,15);

        v16bf bh = *(const v16bf*)(whp + kb);
        v16bf bl = *(const v16bf*)(wlp + kb);

        acc = __builtin_amdgcn_wmma_f32_16x16x32_bf16(false, ah, false, bh,
                                                      (short)0, acc, false, false);
        acc = __builtin_amdgcn_wmma_f32_16x16x32_bf16(false, ah, false, bl,
                                                      (short)0, acc, false, false);
        acc = __builtin_amdgcn_wmma_f32_16x16x32_bf16(false, al, false, bh,
                                                      (short)0, acc, false, false);
        __syncthreads();
    }

    // store: lane holds rows m = r + 8*h, column n of its tile
    if (wave == 0) {
        if (n < R_) {
            float bias = b_reg[g * R_ + n];
            #pragma unroll
            for (int r = 0; r < 8; ++r) {
                int m = r + 8 * h;
                if (t * 16 + m < cnt)
                    out[(size_t)s_rows[m] * R_ + n] = acc[r] + bias;
            }
        }
    } else {
        int nc = (wave - 1) * 16 + n;
        float bias = b_cls[g * KC_ + nc];
        float* ocls = out + (size_t)B_ * R_;
        #pragma unroll
        for (int r = 0; r < 8; ++r) {
            int m = r + 8 * h;
            if (t * 16 + m < cnt)
                ocls[(size_t)s_rows[m] * KC_ + nc] = acc[r] + bias;
        }
    }
}

extern "C" void kernel_launch(void* const* d_in, const int* in_sizes, int n_in,
                              void* d_out, int out_size, void* d_ws, size_t ws_size,
                              hipStream_t stream) {
    const float* feats   = (const float*)d_in[0];
    const int*   roi_ids = (const int*)  d_in[1];
    const float* W_reg   = (const float*)d_in[2];
    const float* b_reg   = (const float*)d_in[3];
    const float* W_cls   = (const float*)d_in[4];
    const float* b_cls   = (const float*)d_in[5];
    float* out = (float*)d_out;

    char* ws = (char*)d_ws;
    int*    counts = (int*)   (ws + WS_COUNTS);
    int*    order  = (int*)   (ws + WS_ORDER);
    __bf16* Whi    = (__bf16*)(ws + WS_WHI);
    __bf16* Wlo    = (__bf16*)(ws + WS_WLO);

    roi_zero_counts<<<1, 32, 0, stream>>>(counts);
    roi_bucket<<<B_ / 256, 256, 0, stream>>>(roi_ids, counts, order);
    roi_convert_w<<<(G_ * NPAD_ * D_) / 256, 256, 0, stream>>>(W_reg, W_cls, Whi, Wlo);
    roi_gemm<<<dim3(G_ * 1024), 128, 0, stream>>>(feats, counts, order,
                                                  Whi, Wlo, b_reg, b_cls, out);
}
